// SparseDCNAttentionLayer_72378788873004
// MI455X (gfx1250) — matrix operand order repaired and checked by reading, not compiled, not run on hardware
//
#include <hip/hip_runtime.h>

#define B_SZ   2048
#define F_SZ   256
#define C_SZ   128
#define D_SZ   64
#define TOPK   32
#define N_SZ   384   // F + C
#define EPS    1e-5f

typedef __attribute__((ext_vector_type(16))) __bf16 v16bf;
typedef __attribute__((ext_vector_type(8)))  float  v8f;
typedef __attribute__((ext_vector_type(4)))  float  v4f;
typedef __attribute__((ext_vector_type(4)))  int    v4i;

// AS-qualified vector types for the async-to-LDS builtin:
// arg0 = global (AS1) v4i*, arg1 = LDS (AS3) v4i*  (per hipcc diagnostic)
typedef __attribute__((address_space(1))) v4i gv4i;
typedef __attribute__((address_space(3))) v4i lv4i;

#if defined(__gfx1250__) && __has_builtin(__builtin_amdgcn_global_load_async_to_lds_b128)
#define USE_ASYNC_LDS 1
#else
#define USE_ASYNC_LDS 0
#endif

// ---------------------------------------------------------------------------
// WMMA helpers: D = A(16x32 bf16) * B(32x16 bf16) + C(f32), wave32.
// A layout (16-bit A 16x32): lane m=lane&15, half=lane>>4;
//   elem e -> K = half*16 + e (same k order as B)
// B layout (rows striped across lanes): lane n=lane&15, half=lane>>4;
//   elem e -> K = half*16 + e, col = n
// C/D: elem r -> row = r + half*8, col = lane&15
// ---------------------------------------------------------------------------
__device__ __forceinline__ v16bf cvt16(const float* f) {
  v16bf r;
#pragma unroll
  for (int e = 0; e < 16; ++e) r[e] = (__bf16)f[e];
  return r;
}

// rowptr points at row m of an [M x K] row-major matrix (contiguous in K)
__device__ __forceinline__ v16bf load_a_frag(const float* __restrict__ rowptr,
                                             int k0, int half) {
  float t[16];
  const int ka = k0 + half * 16;
#pragma unroll
  for (int e = 0; e < 16; ++e) t[e] = rowptr[ka + e];
  return cvt16(t);
}

// colptr points at "row n" of B^T, i.e. B[k][n] contiguous in k
__device__ __forceinline__ v16bf load_b_frag(const float* __restrict__ colptr,
                                             int k0, int half) {
  float t[16];
  const float* p = colptr + k0 + half * 16;
#pragma unroll
  for (int e = 0; e < 16; ++e) t[e] = p[e];
  return cvt16(t);
}

__device__ __forceinline__ v8f wmma_bf16(v16bf a, v16bf b, v8f c) {
  return __builtin_amdgcn_wmma_f32_16x16x32_bf16(
      /*neg_a=*/false, a, /*neg_b=*/false, b,
      /*c_mod=*/(short)0, c, /*reuse_a=*/false, /*reuse_b=*/false);
}

// ---------------------------------------------------------------------------
// Kernel 1: column sums of Wq / Wc  (wqsum[d] = sum_e W[e][d])
// ---------------------------------------------------------------------------
__global__ void sums_kernel(const float* __restrict__ Wq,
                            const float* __restrict__ Wc,
                            float* __restrict__ wqsum,
                            float* __restrict__ wcsum) {
  const int tid = threadIdx.x;
  if (tid < D_SZ) {
    float s = 0.f;
    for (int e = 0; e < D_SZ; ++e) s += Wq[e * D_SZ + tid];
    wqsum[tid] = s;
  } else if (tid < 2 * D_SZ) {
    const int d = tid - D_SZ;
    float s = 0.f;
    for (int e = 0; e < D_SZ; ++e) s += Wc[e * D_SZ + d];
    wcsum[d] = s;
  }
}

// ---------------------------------------------------------------------------
// Kernel 2: x0[b,n] = dot(vec[b,n,:], colsum)  (4 lanes per row, quad reduce)
// ---------------------------------------------------------------------------
__global__ __launch_bounds__(256) void x0_kernel(
    const float* __restrict__ feat, const float* __restrict__ ctx,
    const float* __restrict__ wqsum, const float* __restrict__ wcsum,
    float* __restrict__ x0) {
  const int gid  = blockIdx.x * 256 + threadIdx.x;
  const int row  = gid >> 2;     // (b, n) flat, B*N rows
  const int chunk = gid & 3;     // 16 floats each
  const int b = row / N_SZ;
  const int n = row - b * N_SZ;
  const float* src;
  const float* sv;
  if (n < F_SZ) { src = feat + ((size_t)b * F_SZ + n) * D_SZ;          sv = wqsum; }
  else          { src = ctx  + ((size_t)b * C_SZ + (n - F_SZ)) * D_SZ; sv = wcsum; }
  const int base = chunk * 16;
  float s = 0.f;
#pragma unroll
  for (int t = 0; t < 16; t += 4) {
    float4 a = *(const float4*)(src + base + t);
    float4 w = *(const float4*)(sv + base + t);
    s += a.x * w.x + a.y * w.y + a.z * w.z + a.w * w.w;
  }
  s += __shfl_xor(s, 1);
  s += __shfl_xor(s, 2);
  if (chunk == 0) x0[row] = s;
}

// ---------------------------------------------------------------------------
// Kernel 3: one DCNv2 cross step: xout = x0 * (xin @ W^T + bias) + xin
// One wave per 16x16 output tile, K = 384 in 12 bf16 WMMAs.
// ---------------------------------------------------------------------------
__global__ __launch_bounds__(32) void cross_kernel(
    const float* __restrict__ xin, const float* __restrict__ x0,
    const float* __restrict__ W, const float* __restrict__ bias,
    float* __restrict__ xout) {
  const int lane = threadIdx.x & 31;
  const int tile = blockIdx.x;
  const int tm   = tile / (N_SZ / 16);
  const int tn   = tile % (N_SZ / 16);
  const int m    = lane & 15;
  const int half = lane >> 4;
  const float* arow = xin + (size_t)(tm * 16 + m) * N_SZ;   // A row (M=m, K contiguous)
  const float* bcol = W   + (size_t)(tn * 16 + m) * N_SZ;   // W row j == B^T row (K contiguous)
  v8f acc = {};
#pragma unroll 4
  for (int k0 = 0; k0 < N_SZ; k0 += 32)
    acc = wmma_bf16(load_a_frag(arow, k0, half), load_b_frag(bcol, k0, half), acc);
  const int gc = tn * 16 + m;
  const float bn = bias[gc];
#pragma unroll
  for (int r = 0; r < 8; ++r) {
    const int gr = tm * 16 + half * 8 + r;
    const size_t off = (size_t)gr * N_SZ + gc;
    xout[off] = x0[off] * (acc[r] + bn) + xin[off];
  }
}

// ---------------------------------------------------------------------------
// Kernel 4: per-batch stable top-k (32 of 256 feature slots, 32 of 128 ctx)
// Iterative argmax with smallest-index tie-break (matches lax.top_k order).
// ---------------------------------------------------------------------------
__global__ __launch_bounds__(256) void topk_kernel(
    const float* __restrict__ wgt, int* __restrict__ idxF, int* __restrict__ idxC) {
  __shared__ float sv[N_SZ];
  __shared__ float rv[256];
  __shared__ int   ri[256];
  const int b = blockIdx.x;
  const int tid = threadIdx.x;
  for (int t = tid; t < N_SZ; t += 256) sv[t] = wgt[(size_t)b * N_SZ + t];
  __syncthreads();
  // feature part: 256 candidates
  for (int it = 0; it < TOPK; ++it) {
    rv[tid] = sv[tid]; ri[tid] = tid;
    __syncthreads();
    for (int s = 128; s > 0; s >>= 1) {
      if (tid < s) {
        float v2 = rv[tid + s]; int i2 = ri[tid + s];
        if (v2 > rv[tid] || (v2 == rv[tid] && i2 < ri[tid])) { rv[tid] = v2; ri[tid] = i2; }
      }
      __syncthreads();
    }
    if (tid == 0) { idxF[b * TOPK + it] = ri[0]; sv[ri[0]] = -__builtin_inff(); }
    __syncthreads();
  }
  // context part: 128 candidates
  for (int it = 0; it < TOPK; ++it) {
    if (tid < 128) { rv[tid] = sv[F_SZ + tid]; ri[tid] = tid; }
    __syncthreads();
    for (int s = 64; s > 0; s >>= 1) {
      if (tid < s) {
        float v2 = rv[tid + s]; int i2 = ri[tid + s];
        if (v2 > rv[tid] || (v2 == rv[tid] && i2 < ri[tid])) { rv[tid] = v2; ri[tid] = i2; }
      }
      __syncthreads();
    }
    if (tid == 0) { idxC[b * TOPK + it] = ri[0]; sv[F_SZ + ri[0]] = -__builtin_inff(); }
    __syncthreads();
  }
}

// ---------------------------------------------------------------------------
// Kernel 5 (fused, memory-bound stage): per batch b —
//   async-gather top-k rows to LDS -> tf = rows@Wq^T, tc = rows@Wc^T (WMMA) ->
//   stats via WMMA GEMMs (mu = tf@tc^T/64, E[v^2] = tf^2@tc2^T/64) ->
//   normalized outer product, wave-coalesced b128 NT stores (256 KB/block).
// ---------------------------------------------------------------------------
__global__ __launch_bounds__(256) void output_kernel(
    const float* __restrict__ feat, const float* __restrict__ ctx,
    const float* __restrict__ Wq, const float* __restrict__ Wc,
    const float* __restrict__ gamma, const float* __restrict__ beta,
    const int* __restrict__ idxF, const int* __restrict__ idxC,
    float* __restrict__ out) {
  __shared__ __align__(16) float s_fraw[TOPK * D_SZ];
  __shared__ __align__(16) float s_craw[TOPK * D_SZ];
  __shared__ __align__(16) float s_tf [TOPK * D_SZ];
  __shared__ __align__(16) float s_tc [TOPK * D_SZ];
  __shared__ __align__(16) float s_tf2[TOPK * D_SZ];
  __shared__ __align__(16) float s_tc2[TOPK * D_SZ];
  __shared__ __align__(16) float s_s1 [TOPK * TOPK];
  __shared__ __align__(16) float s_s2 [TOPK * TOPK];
  __shared__ __align__(16) float s_g[D_SZ];
  __shared__ __align__(16) float s_b[D_SZ];

  const int b    = blockIdx.x;
  const int tid  = threadIdx.x;
  const int wave = tid >> 5;
  const int lane = tid & 31;
  const int m    = lane & 15;
  const int half = lane >> 4;

  // phase 0: gather selected rows into LDS (async-to-LDS path when available)
  for (int t = tid; t < TOPK * (D_SZ / 4); t += 256) {  // 512 float4 each
    const int rowi = t >> 4;
    const int c4   = t & 15;
    const int srcF = idxF[b * TOPK + rowi];
    const int srcC = idxC[b * TOPK + rowi];
    const float* gf = feat + ((size_t)b * F_SZ + srcF) * D_SZ + c4 * 4;
    const float* gx = ctx  + ((size_t)b * C_SZ + srcC) * D_SZ + c4 * 4;
#if USE_ASYNC_LDS
    __builtin_amdgcn_global_load_async_to_lds_b128(
        (gv4i*)gf, (lv4i*)&s_fraw[t * 4], 0, 0);
    __builtin_amdgcn_global_load_async_to_lds_b128(
        (gv4i*)gx, (lv4i*)&s_craw[t * 4], 0, 0);
#else
    ((float4*)s_fraw)[t] = *(const float4*)gf;
    ((float4*)s_craw)[t] = *(const float4*)gx;
#endif
  }
  if (tid < D_SZ) { s_g[tid] = gamma[tid]; s_b[tid] = beta[tid]; }
#if USE_ASYNC_LDS
#if __has_builtin(__builtin_amdgcn_s_wait_asynccnt)
  __builtin_amdgcn_s_wait_asynccnt(0);
#else
  asm volatile("s_wait_asynccnt 0" ::: "memory");
#endif
#endif
  __syncthreads();

  // phase 1: tf = fraw @ Wq^T, tc = craw @ Wc^T  (16 tiles over 8 waves)
  for (int t = wave; t < 16; t += 8) {
    const int matsel = t >> 3;        // 0 = tf, 1 = tc
    const int t2 = t & 7;
    const int tm = t2 >> 2;           // 0..1 (row tile of 32)
    const int tn = t2 & 3;            // 0..3 (col tile of 64)
    const float* Araw = matsel ? s_craw : s_fraw;
    const float* Wt   = matsel ? Wc : Wq;          // W[e][d]: row e contiguous in d == B^T
    float* dst  = matsel ? s_tc  : s_tf;
    float* dst2 = matsel ? s_tc2 : s_tf2;
    const float* arow = Araw + (size_t)(tm * 16 + m) * D_SZ;
    const float* bcol = Wt   + (size_t)(tn * 16 + m) * D_SZ;
    v8f acc = {};
#pragma unroll
    for (int k0 = 0; k0 < D_SZ; k0 += 32)
      acc = wmma_bf16(load_a_frag(arow, k0, half), load_b_frag(bcol, k0, half), acc);
#pragma unroll
    for (int r = 0; r < 8; ++r) {
      const int gr = tm * 16 + half * 8 + r;
      const int gc = tn * 16 + m;
      const float v = acc[r];
      dst [gr * D_SZ + gc] = v;
      dst2[gr * D_SZ + gc] = v * v;
    }
  }
  __syncthreads();

  // phase 2: s1 = tf @ tc^T, s2 = tf2 @ tc2^T  (8 tiles, one per wave)
  {
    const int sel = wave >> 2;
    const int t2  = wave & 3;
    const int tm  = t2 >> 1;
    const int tn  = t2 & 1;
    const float* A  = sel ? s_tf2 : s_tf;
    const float* Bt = sel ? s_tc2 : s_tc;          // tc[j][d] contiguous in d == B^T
    float* dst = sel ? s_s2 : s_s1;
    const float* arow = A  + (size_t)(tm * 16 + m) * D_SZ;
    const float* bcol = Bt + (size_t)(tn * 16 + m) * D_SZ;
    v8f acc = {};
#pragma unroll
    for (int k0 = 0; k0 < D_SZ; k0 += 32)
      acc = wmma_bf16(load_a_frag(arow, k0, half), load_b_frag(bcol, k0, half), acc);
#pragma unroll
    for (int r = 0; r < 8; ++r)
      dst[(tm * 16 + half * 8 + r) * TOPK + tn * 16 + m] = acc[r];
  }
  __syncthreads();

  // phase 3: normalized outer product; wave handles 2 pairs/iter,
  // 16 lanes x b128 per pair row -> 512B contiguous NT store per instruction.
  const int sub = lane >> 4;          // which of the 2 pairs this half-wave owns
  const int d0  = (lane & 15) * 4;    // 4 consecutive d per lane
  const v4f g4 = *(const v4f*)&s_g[d0];
  const v4f b4 = *(const v4f*)&s_b[d0];
  float* outb = out + (size_t)b * (TOPK * TOPK * D_SZ);
  for (int p0 = wave * 128; p0 < wave * 128 + 128; p0 += 2) {
    const int p = p0 + sub;
    const int i = p >> 5, j = p & 31;
    const float mu  = s_s1[p] * (1.0f / D_SZ);
    const float var = s_s2[p] * (1.0f / D_SZ) - mu * mu;
    const float rs  = __frsqrt_rn(var + EPS);
    const v4f tf4 = *(const v4f*)&s_tf[i * D_SZ + d0];
    const v4f tc4 = *(const v4f*)&s_tc[j * D_SZ + d0];
    v4f o;
    o.x = (tf4.x * tc4.x - mu) * rs * g4.x + b4.x;
    o.y = (tf4.y * tc4.y - mu) * rs * g4.y + b4.y;
    o.z = (tf4.z * tc4.z - mu) * rs * g4.z + b4.z;
    o.w = (tf4.w * tc4.w - mu) * rs * g4.w + b4.w;
    __builtin_nontemporal_store(o, (v4f*)(outb + (size_t)p * D_SZ + d0));
  }
}

// ---------------------------------------------------------------------------
extern "C" void kernel_launch(void* const* d_in, const int* in_sizes, int n_in,
                              void* d_out, int out_size, void* d_ws, size_t ws_size,
                              hipStream_t stream) {
  (void)in_sizes; (void)n_in; (void)out_size; (void)ws_size;
  const float* feat   = (const float*)d_in[0];
  const float* ctx    = (const float*)d_in[1];
  const float* Wq     = (const float*)d_in[2];
  const float* Wc     = (const float*)d_in[3];
  const float* crossW = (const float*)d_in[4];   // [3, 384, 384]
  const float* crossB = (const float*)d_in[5];   // [3, 384]
  const float* gamma  = (const float*)d_in[6];
  const float* beta   = (const float*)d_in[7];
  float* out = (float*)d_out;

  char* p = (char*)d_ws;
  float* wqsum = (float*)p; p += 64 * sizeof(float);
  float* wcsum = (float*)p; p += 64 * sizeof(float);
  float* x0 = (float*)p; p += (size_t)B_SZ * N_SZ * sizeof(float);
  float* xa = (float*)p; p += (size_t)B_SZ * N_SZ * sizeof(float);
  float* xb = (float*)p; p += (size_t)B_SZ * N_SZ * sizeof(float);
  int* idxF = (int*)p; p += (size_t)B_SZ * TOPK * sizeof(int);
  int* idxC = (int*)p; p += (size_t)B_SZ * TOPK * sizeof(int);

  sums_kernel<<<1, 128, 0, stream>>>(Wq, Wc, wqsum, wcsum);
  x0_kernel<<<(B_SZ * N_SZ * 4) / 256, 256, 0, stream>>>(feat, ctx, wqsum, wcsum, x0);

  const int cgrid = (B_SZ / 16) * (N_SZ / 16);
  cross_kernel<<<cgrid, 32, 0, stream>>>(x0, x0, crossW, crossB, xa);
  cross_kernel<<<cgrid, 32, 0, stream>>>(xa, x0, crossW + (size_t)N_SZ * N_SZ, crossB + N_SZ, xb);
  cross_kernel<<<cgrid, 32, 0, stream>>>(xb, x0, crossW + (size_t)2 * N_SZ * N_SZ, crossB + 2 * N_SZ, xa);

  topk_kernel<<<B_SZ, 256, 0, stream>>>(xa, idxF, idxC);
  output_kernel<<<B_SZ, 256, 0, stream>>>(feat, ctx, Wq, Wc, gamma, beta, idxF, idxC, out);
}
